// SiTBlockWithDINOKV_11785390260606
// MI455X (gfx1250) — matrix operand order repaired and checked by reading, not compiled, not run on hardware
//
#include <hip/hip_runtime.h>
#include <cmath>

typedef _Float16 v16h __attribute__((ext_vector_type(16)));
typedef _Float16 v8h  __attribute__((ext_vector_type(8)));
typedef _Float16 v4h  __attribute__((ext_vector_type(4)));
typedef float    v8f  __attribute__((ext_vector_type(8)));
typedef float    v4f  __attribute__((ext_vector_type(4)));

union AF { v16h v; v8h h[2]; _Float16 e[16]; };

__device__ __forceinline__ v8f wmma_f16(v16h a, v16h b, v8f c){
  // (neg_a, A, neg_b, B, c_mod, C, reuse_a, reuse_b)
  return __builtin_amdgcn_wmma_f32_16x16x32_f16(false, a, false, b, (short)0, c, false, false);
}

// CDNA5 async global->LDS copy (tracked by ASYNCcnt, no VGPR data movement).
// lds_off = workgroup-relative LDS byte address (low 32 bits of flat pointer).
__device__ __forceinline__ void async_b128(void* lds_ptr, const void* gptr){
  unsigned lds_off = (unsigned)(size_t)lds_ptr;
  unsigned long long ga = (unsigned long long)(size_t)gptr;
  asm volatile("global_load_async_to_lds_b128 %0, %1, off"
               :: "v"(lds_off), "v"(ga) : "memory");
}
__device__ __forceinline__ void wait_async0(){
  asm volatile("s_wait_asynccnt 0" ::: "memory");
}

#define NN 1024
#define HD 72

// ---------------------------------------------------------------------------
// fp32 -> f16 bulk convert (vectorized x4). n4 = n/4.
// ---------------------------------------------------------------------------
__global__ __launch_bounds__(256) void cvt_f16(
    const float* __restrict__ in, _Float16* __restrict__ out, int n4)
{
  int i = blockIdx.x * 256 + threadIdx.x;
  if (i < n4) {
    v4f f = *(const v4f*)(in + 4 * (size_t)i);
    v4h h;
    for (int k = 0; k < 4; k++) h[k] = (_Float16)f[k];
    *(v4h*)(out + 4 * (size_t)i) = h;
  }
}

// ---------------------------------------------------------------------------
// C[M x Nout] = A[M x K] * W[Nout x K]^T + bias.  A, W are f16; C f16 or f32.
// Block tile 32(M) x 64(N), 128 threads = 4 waves, K-step 64 (2 WMMA chunks).
// Tile staging uses CDNA5 async global->LDS b128 copies.
// ---------------------------------------------------------------------------
template<bool OUT16>
__global__ __launch_bounds__(128) void gemm_h(
    const _Float16* __restrict__ A, const _Float16* __restrict__ W,
    const float* __restrict__ bias, void* __restrict__ Cout,
    int K, int Nout)
{
  __shared__ __align__(16) _Float16 As[32][80];
  __shared__ __align__(16) _Float16 Bs[64][80];
  const int tid  = threadIdx.x;
  const int lane = tid & 31;
  const int w    = tid >> 5;
  const int mi   = w & 1;          // M sub-tile
  const int nt0  = (w >> 1) * 2;   // first of 2 N sub-tiles
  const int gm0  = blockIdx.y * 32;
  const int gn0  = blockIdx.x * 64;

  v8f acc[2];
  for (int t = 0; t < 2; t++) for (int i = 0; i < 8; i++) acc[t][i] = 0.f;

  const int ar  = tid >> 2, ac0 = (tid & 3) * 16;  // A: 16 halves/thread
  const int br  = tid >> 1, bc0 = (tid & 1) * 32;  // B: 32 halves/thread

  for (int kk = 0; kk < K; kk += 64) {
    { // stage A tile 32x64 halves: async global->LDS, no VGPR round-trip
      const _Float16* src = A + (size_t)(gm0 + ar) * K + kk + ac0;
      async_b128(&As[ar][ac0],     src);
      async_b128(&As[ar][ac0 + 8], src + 8);
    }
    { // stage B tile 64x64 halves: rows of W are transposed-B columns directly
      const _Float16* src = W + (size_t)(gn0 + br) * K + kk + bc0;
      async_b128(&Bs[br][bc0],      src);
      async_b128(&Bs[br][bc0 + 8],  src + 8);
      async_b128(&Bs[br][bc0 + 16], src + 16);
      async_b128(&Bs[br][bc0 + 24], src + 24);
    }
    wait_async0();        // ASYNCcnt == 0: this wave's copies landed in LDS
    __syncthreads();      // all waves' copies visible

    for (int ck = 0; ck < 2; ck++) {
      const int kb = ck * 32;
      AF ua; // A frag: lane<16 -> K kb+0-7 & kb+16-23 of row M=lane; lane>=16: +8
      const int arow = mi * 16 + (lane & 15);
      const int acol = kb + (lane >> 4) * 8;
      ua.h[0] = *(const v8h*)&As[arow][acol];
      ua.h[1] = *(const v8h*)&As[arow][acol + 16];
      for (int t = 0; t < 2; t++) {
        AF ub; // B frag: lane<16 -> K kb+0-7 & kb+16-23 of col N=lane; lane>=16: +8
        const int brow = (nt0 + t) * 16 + (lane & 15);
        const int bcol = kb + (lane >> 4) * 8;
        ub.h[0] = *(const v8h*)&Bs[brow][bcol];
        ub.h[1] = *(const v8h*)&Bs[brow][bcol + 16];
        acc[t] = wmma_f16(ua.v, ub.v, acc[t]);
      }
    }
    __syncthreads();
  }

  for (int t = 0; t < 2; t++) {
    const int col = gn0 + (nt0 + t) * 16 + (lane & 15);
    const float bv = bias[col];
    for (int vv = 0; vv < 8; vv++) {
      const int row = gm0 + mi * 16 + vv + 8 * (lane >> 4);
      const float val = acc[t][vv] + bv;
      if (OUT16) ((_Float16*)Cout)[(size_t)row * Nout + col] = (_Float16)val;
      else       ((float*)Cout)[(size_t)row * Nout + col]    = val;
    }
  }
}

// ---------------------------------------------------------------------------
// Fused dual attention + alignment losses. One block = (b, h, 16 query rows).
// f16 logits rows (SiT & DINO) live in LDS; V chunks staged transposed in LDS.
// ---------------------------------------------------------------------------
__global__ __launch_bounds__(256) void attn_fused(
    const _Float16* __restrict__ qkv,   // [2048][3456] f16 : q | k_sit | v_sit
    const _Float16* __restrict__ kd,    // [2][16][1024][72] f16
    const _Float16* __restrict__ vd,    // [2][16][1024][72] f16
    _Float16* __restrict__ attn_out,    // [2048][1152] f16
    float* __restrict__ lpart, float* __restrict__ apart)
{
  __shared__ __align__(16) _Float16 Shs[16][1024];   // 32 KB SiT logits/probs
  __shared__ __align__(16) _Float16 Shd[16][1024];   // 32 KB DINO logits/probs
  __shared__ __align__(16) _Float16 Vt[2][80][32];   // 10 KB V^T chunk (sit,dino)
  __shared__ __align__(16) float Ats[16][80];
  __shared__ __align__(16) float Atd[16][80];
  __shared__ float inv_s[16], inv_d[16], lrow[16], arow_[16];

  const int bid = blockIdx.x;
  const int b  = bid >> 10;
  const int h  = (bid >> 6) & 15;
  const int n0 = (bid & 63) << 4;
  const int tid  = threadIdx.x;
  const int lane = tid & 31;
  const int w    = tid >> 5;
  const float scale = 0.11785113019775792f; // 72^-0.5
  const v8h zero8 = (v8h)(_Float16)0.f;

  const size_t dino_base = ((size_t)(b * 16 + h)) * NN * HD;

  // ---- Phase 1: logits = q . k^T * scale (K padded 72->96, guards on whole
  //      8-half chunks since 72 % 8 == 0) ----
  AF aq[3];
  {
    const _Float16* qp = qkv + (size_t)(b * NN + n0 + (lane & 15)) * 3456 + h * HD;
    for (int c = 0; c < 3; c++) {
      const int kb = 32 * c + (lane >> 4) * 8;
      aq[c].h[0] = (kb + 8  <= HD) ? *(const v8h*)(qp + kb)      : zero8;
      aq[c].h[1] = (kb + 24 <= HD) ? *(const v8h*)(qp + kb + 16) : zero8;
    }
  }
  for (int nt = w * 8; nt < w * 8 + 8; nt++) {   // 8 waves x 8 key tiles = 1024
    const int nk = nt * 16 + (lane & 15);
    const _Float16* kps = qkv + (size_t)(b * NN + nk) * 3456 + 1152 + h * HD;
    const _Float16* kpd = kd + dino_base + (size_t)nk * HD;
    v8f acs, acd;
    for (int i = 0; i < 8; i++) { acs[i] = 0.f; acd[i] = 0.f; }
    for (int c = 0; c < 3; c++) {
      const int kb = 32 * c + (lane >> 4) * 8;
      AF ubs, ubd;
      ubs.h[0] = (kb + 8  <= HD) ? *(const v8h*)(kps + kb)      : zero8;
      ubs.h[1] = (kb + 24 <= HD) ? *(const v8h*)(kps + kb + 16) : zero8;
      ubd.h[0] = (kb + 8  <= HD) ? *(const v8h*)(kpd + kb)      : zero8;
      ubd.h[1] = (kb + 24 <= HD) ? *(const v8h*)(kpd + kb + 16) : zero8;
      acs = wmma_f16(aq[c].v, ubs.v, acs);
      acd = wmma_f16(aq[c].v, ubd.v, acd);
    }
    const int cc = nt * 16 + (lane & 15);
    for (int vv = 0; vv < 8; vv++) {
      const int r = vv + 8 * (lane >> 4);
      Shs[r][cc] = (_Float16)(acs[vv] * scale);
      Shd[r][cc] = (_Float16)(acd[vv] * scale);
    }
  }
  __syncthreads();

  // ---- Phase 2: per-row max / L2 / dot, then softmax exp in-place (f16) ----
  {
    const int r = tid >> 4, sub = tid & 15;   // 16 lanes per row, same wave
    float mxs = -3.4e38f, mxd = -3.4e38f, sss = 0.f, ssd = 0.f, sdot = 0.f;
    for (int j = sub; j < 1024; j += 16) {
      float a = (float)Shs[r][j], d = (float)Shd[r][j];
      mxs = fmaxf(mxs, a); mxd = fmaxf(mxd, d);
      sss += a * a; ssd += d * d; sdot += a * d;
    }
    for (int m = 8; m; m >>= 1) {
      mxs = fmaxf(mxs, __shfl_xor(mxs, m, 16));
      mxd = fmaxf(mxd, __shfl_xor(mxd, m, 16));
      sss  += __shfl_xor(sss,  m, 16);
      ssd  += __shfl_xor(ssd,  m, 16);
      sdot += __shfl_xor(sdot, m, 16);
    }
    float es = 0.f, ed = 0.f;
    for (int j = sub; j < 1024; j += 16) {
      _Float16 a = (_Float16)__expf((float)Shs[r][j] - mxs);
      _Float16 d = (_Float16)__expf((float)Shd[r][j] - mxd);
      Shs[r][j] = a; es += (float)a;
      Shd[r][j] = d; ed += (float)d;
    }
    for (int m = 8; m; m >>= 1) { es += __shfl_xor(es, m, 16); ed += __shfl_xor(ed, m, 16); }
    if (sub == 0) {
      inv_s[r] = 1.f / es; inv_d[r] = 1.f / ed;
      float ns = fmaxf(sqrtf(sss), 1e-12f);
      float nd = fmaxf(sqrtf(ssd), 1e-12f);
      lrow[r] = sdot / (ns * nd);   // cosine of logits rows
    }
  }
  __syncthreads();

  // ---- Phase 3: attn = softmax(P) @ V, chunk-synchronized V staging ----
  // 10 jobs = {sit,dino} x 5 d-tiles; wave w owns jobs {w, w+8}.
  const int njob = (w < 2) ? 2 : 1;
  int jobs[2] = { w, w + 8 };
  v8f acc[2];
  for (int j = 0; j < 2; j++) for (int i = 0; i < 8; i++) acc[j][i] = 0.f;

  for (int kc = 0; kc < 32; kc++) {
    const int kbase = kc * 32;
    __syncthreads();   // previous chunk's Vt reads complete
    // stage V^T chunk: Vt[mat][d][kk] = V[kbase+kk][d] (d>=72 zero-filled)
    for (int i = tid; i < 2 * 2560; i += 256) {
      const int mat = (i >= 2560);
      const int r2  = i - mat * 2560;
      const int kk  = r2 / 80;
      const int d   = r2 - kk * 80;
      _Float16 val = (_Float16)0.f;
      if (d < HD) {
        val = mat ? vd[dino_base + (size_t)(kbase + kk) * HD + d]
                  : qkv[(size_t)(b * NN + kbase + kk) * 3456 + 2304 + h * HD + d];
      }
      Vt[mat][d][kk] = val;
    }
    __syncthreads();

    for (int j = 0; j < njob; j++) {
      const int job = jobs[j];
      const int mat = job & 1;
      const int d0  = (job >> 1) * 16;
      _Float16 (*Sm)[1024] = mat ? Shd : Shs;
      AF ua;
      {
        const int arowi = lane & 15;
        const int ac = kbase + (lane >> 4) * 8;
        ua.h[0] = *(const v8h*)&Sm[arowi][ac];
        ua.h[1] = *(const v8h*)&Sm[arowi][ac + 16];
      }
      AF ub;
      {
        const int dcol = d0 + (lane & 15);
        const int kb2  = (lane >> 4) * 8;
        ub.h[0] = *(const v8h*)&Vt[mat][dcol][kb2];
        ub.h[1] = *(const v8h*)&Vt[mat][dcol][kb2 + 16];
      }
      acc[j] = wmma_f16(ua.v, ub.v, acc[j]);
    }
  }

  for (int j = 0; j < njob; j++) {
    const int job = jobs[j];
    const int mat = job & 1;
    const int dcol = (job >> 1) * 16 + (lane & 15);
    for (int vv = 0; vv < 8; vv++) {
      const int r = vv + 8 * (lane >> 4);
      const float iv  = mat ? inv_d[r] : inv_s[r];
      const float val = acc[j][vv] * iv;
      if (mat) Atd[r][dcol] = val; else Ats[r][dcol] = val;
      if (!mat && dcol < HD)
        attn_out[(size_t)(b * NN + n0 + r) * 1152 + h * HD + dcol] = (_Float16)val;
    }
  }
  __syncthreads();

  // ---- Phase 4: attention-row cosine (len 72) ----
  if (tid < 128) {
    const int r = tid >> 3, sub = tid & 7;
    float ss = 0.f, dd = 0.f, cr = 0.f;
    for (int j = sub; j < HD; j += 8) {
      float a = Ats[r][j], d2 = Atd[r][j];
      ss += a * a; dd += d2 * d2; cr += a * d2;
    }
    for (int m = 4; m; m >>= 1) {
      ss += __shfl_xor(ss, m, 8);
      dd += __shfl_xor(dd, m, 8);
      cr += __shfl_xor(cr, m, 8);
    }
    if (sub == 0)
      arow_[r] = cr / (fmaxf(sqrtf(ss), 1e-12f) * fmaxf(sqrtf(dd), 1e-12f));
  }
  __syncthreads();
  if (tid == 0) {
    float L = 0.f, A = 0.f;
    for (int r = 0; r < 16; r++) { L += lrow[r]; A += arow_[r]; }
    lpart[bid] = L; apart[bid] = A;
  }
}

// ---------------------------------------------------------------------------
// Deterministic fixed-order reduction of 2048 per-block partials -> loss.
// ---------------------------------------------------------------------------
__global__ __launch_bounds__(256) void loss_reduce(
    const float* __restrict__ lpart, const float* __restrict__ apart,
    float* __restrict__ out_loss)
{
  __shared__ float sl[256], sa[256];
  float l = 0.f, a = 0.f;
  for (int i = threadIdx.x; i < 2048; i += 256) { l += lpart[i]; a += apart[i]; }
  sl[threadIdx.x] = l; sa[threadIdx.x] = a;
  __syncthreads();
  for (int s = 128; s > 0; s >>= 1) {
    if ((int)threadIdx.x < s) {
      sl[threadIdx.x] += sl[threadIdx.x + s];
      sa[threadIdx.x] += sa[threadIdx.x + s];
    }
    __syncthreads();
  }
  if (threadIdx.x == 0) {
    const float inv_rows = 1.f / 32768.f;   // B*H*N
    float logit_align = 1.f - sl[0] * inv_rows;
    float attn_align  = 1.f - sa[0] * inv_rows;
    *out_loss = logit_align + 0.5f * attn_align;
  }
}

extern "C" void kernel_launch(void* const* d_in, const int* in_sizes, int n_in,
                              void* d_out, int out_size, void* d_ws, size_t ws_size,
                              hipStream_t stream)
{
  const float* x      = (const float*)d_in[0];
  const float* qkv_w  = (const float*)d_in[1];
  const float* qkv_b  = (const float*)d_in[2];
  const float* proj_w = (const float*)d_in[3];
  const float* proj_b = (const float*)d_in[4];
  const float* k_dino = (const float*)d_in[5];
  const float* v_dino = (const float*)d_in[6];
  (void)in_sizes; (void)n_in; (void)out_size; (void)ws_size;

  float* out = (float*)d_out;

  // workspace layout: f16 copies | f16 intermediates | f32 partials
  _Float16* x16   = (_Float16*)d_ws;
  _Float16* qw16  = x16   + (size_t)2048 * 1152;
  _Float16* pw16  = qw16  + (size_t)3456 * 1152;
  _Float16* kd16  = pw16  + (size_t)1152 * 1152;
  _Float16* vd16  = kd16  + (size_t)2359296;
  _Float16* qkv16 = vd16  + (size_t)2359296;
  _Float16* at16  = qkv16 + (size_t)2048 * 3456;
  float*    lpart = (float*)(at16 + (size_t)2048 * 1152);
  float*    apart = lpart + 2048;

  // one-time fp32 -> f16 copies (bandwidth-trivial at 23.3 TB/s)
  cvt_f16<<<dim3((2048 * 1152 / 4 + 255) / 256), 256, 0, stream>>>(x,      x16,  2048 * 1152 / 4);
  cvt_f16<<<dim3((3456 * 1152 / 4 + 255) / 256), 256, 0, stream>>>(qkv_w,  qw16, 3456 * 1152 / 4);
  cvt_f16<<<dim3((1152 * 1152 / 4 + 255) / 256), 256, 0, stream>>>(proj_w, pw16, 1152 * 1152 / 4);
  cvt_f16<<<dim3((2359296 / 4 + 255) / 256),     256, 0, stream>>>(k_dino, kd16, 2359296 / 4);
  cvt_f16<<<dim3((2359296 / 4 + 255) / 256),     256, 0, stream>>>(v_dino, vd16, 2359296 / 4);

  // qkv = x @ qkv_w^T + qkv_b   (f16 out)      M=2048 K=1152 N=3456
  gemm_h<true><<<dim3(3456 / 64, 2048 / 32), 128, 0, stream>>>(
      x16, qw16, qkv_b, (void*)qkv16, 1152, 3456);

  // fused dual attention + losses              2048 blocks = B*H*(N/16)
  attn_fused<<<dim3(2048), 256, 0, stream>>>(qkv16, kd16, vd16, at16, lpart, apart);

  // out = attn_sit @ proj_w^T + proj_b (f32)   M=2048 K=1152 N=1152
  gemm_h<false><<<dim3(1152 / 64, 2048 / 32), 128, 0, stream>>>(
      at16, pw16, proj_b, (void*)out, 1152, 1152);

  // distill loss -> out[2048*1152]
  loss_reduce<<<dim3(1), 256, 0, stream>>>(lpart, apart, out + (size_t)2048 * 1152);
}
